// EncoderSelfAttention_87342454931972
// MI455X (gfx1250) — matrix operand order repaired and checked by reading, not compiled, not run on hardware
//
#include <hip/hip_runtime.h>

// ---------------- problem constants (fixed by the reference) ----------------
#define BB    8
#define SS    1025
#define DD    768
#define NH    12
#define NKV   4
#define HDIM  64
#define SPAD  1056          // S padded to multiple of 32
#define NQKV  1280          // H*HD + 2*KV*HD
#define NREP  3             // NH / NKV
#define GH_C  32
#define GW_C  32
#define EPSC  1e-6f
#define NKT   (SPAD / 16)   // 66 key tiles (even)

typedef __bf16        v16bf __attribute__((ext_vector_type(16)));
typedef float         v8f   __attribute__((ext_vector_type(8)));
typedef unsigned int  v8ui  __attribute__((ext_vector_type(8)));
typedef unsigned int  u32x4 __attribute__((ext_vector_type(4)));
typedef int           i32x4 __attribute__((ext_vector_type(4)));
typedef int           i32x8 __attribute__((ext_vector_type(8)));

// fp32 -> bf16 round-to-nearest-even
__device__ __forceinline__ unsigned short f2bf(float f) {
  unsigned int u = __builtin_bit_cast(unsigned int, f);
  u += 0x7FFFu + ((u >> 16) & 1u);
  return (unsigned short)(u >> 16);
}

// ---------------- WMMA fragment loaders (ISA 7.12.2 layouts, wave32) --------
// Loaders compute the full divergent *byte* offset in 32-bit and add it to a
// uniform base pointer: matches the GVS addressing mode (SGPR base + VGPR i32
// offset), avoiding 64-bit per-lane address math.

// A-matrix 16x32 bf16: lane m=lane&15 holds row m; half=lane>>4 selects
// K pairs {0,2,4,6}+8*half and {16,18,20,22}+8*half => two b128 loads at
// byte offsets half*16 and 32+half*16 from the row's k-base.
__device__ __forceinline__ v16bf load_a_frag(const unsigned short* base, int elem_off, int half) {
  int ob = elem_off * 2 + half * 16;
  uint4 x = *(const uint4*)((const char*)base + ob);
  uint4 y = *(const uint4*)((const char*)base + ob + 32);
  v8ui r;
  r[0] = x.x; r[1] = x.y; r[2] = x.z; r[3] = x.w;
  r[4] = y.x; r[5] = y.y; r[6] = y.z; r[7] = y.w;
  return __builtin_bit_cast(v16bf, r);
}

// B-matrix 32x16 bf16: lane holds column n=lane&15; half=lane>>4 holds
// K=8*half..8*half+7 and 16+8*half..16+8*half+7 from B^T storage (row n).
__device__ __forceinline__ v16bf load_b_frag(const unsigned short* base, int elem_off,
                                             int row_stride, int lane) {
  int n = lane & 15;
  int koff = (lane >> 4) * 8;
  int ob = (elem_off + n * row_stride + koff) * 2;
  uint4 x = *(const uint4*)((const char*)base + ob);
  uint4 y = *(const uint4*)((const char*)base + ob + 32);
  v8ui r;
  r[0] = x.x; r[1] = x.y; r[2] = x.z; r[3] = x.w;
  r[4] = y.x; r[5] = y.y; r[6] = y.z; r[7] = y.w;
  return __builtin_bit_cast(v16bf, r);
}

__device__ __forceinline__ v8f wmma_bf16(v16bf a, v16bf b, v8f c) {
  // (neg_a, A, neg_b, B, c_mod, C, reuse_a, reuse_b)
  return __builtin_amdgcn_wmma_f32_16x16x32_bf16(false, a, false, b, (short)0, c, false, false);
}

// ---------------- Tensor Data Mover: contiguous 2KB tile -> LDS -------------
// D# per CDNA5 ISA: 1 descriptor, type=2, data_size=2B,
// single row tile of 1024 elements (tile_dim0=1024, tile_dim1=1).
__device__ __forceinline__ void tdm_load_2kb(const void* gsrc, void* lds_dst) {
  unsigned long long ga = (unsigned long long)gsrc;
  unsigned int la = (unsigned int)(unsigned long long)lds_dst; // low 32 = LDS offset
  u32x4 g0;
  g0[0] = 1u;                                                  // count=1, user mode
  g0[1] = la;                                                  // lds_addr
  g0[2] = (unsigned int)ga;                                    // global_addr[31:0]
  g0[3] = (unsigned int)((ga >> 32) & 0x01FFFFFFu) | (2u << 30); // addr[56:32] | type=2
  i32x8 g1;
  g1[0] = (1 << 16);               // data_size=1 (2 bytes); no mask/pad/iterate
  g1[1] = (int)(1024u << 16);      // tensor_dim0[15:0] in bits [31:16]
  g1[2] = (int)(1u << 16);         // tensor_dim0[31:16]=0 | tensor_dim1[15:0]=1
  g1[3] = (int)(1024u << 16);      // tensor_dim1[31:16]=0 | tile_dim0=1024
  g1[4] = 1;                       // tile_dim1=1, tile_dim2=0
  g1[5] = 1024;                    // tensor_dim0_stride[31:0]
  g1[6] = 0;                       // stride0 hi | stride1 lo
  g1[7] = 0;                       // stride1 hi
  i32x4 z4 = {};
#if defined(__clang_major__) && __clang_major__ >= 23
  i32x8 z8 = {};
  __builtin_amdgcn_tensor_load_to_lds(g0, g1, z4, z4, z8, 0);
#else
  __builtin_amdgcn_tensor_load_to_lds(g0, g1, z4, z4, 0);
#endif
}

// ---------------- conversion / layout kernels -------------------------------
__global__ void __launch_bounds__(256) cvt_bf16_kernel(const float* __restrict__ src,
                                                       unsigned short* __restrict__ dst, int n) {
  int i = blockIdx.x * blockDim.x + threadIdx.x;
  if (i < n) dst[i] = f2bf(src[i]);
}

// Wcat^T (1280 x 768): row n<768 -> Wq col n; 768..1023 -> Wk; 1024.. -> Wv
__global__ void __launch_bounds__(256) build_wcat_kernel(const float* __restrict__ Wq,
                                                         const float* __restrict__ Wk,
                                                         const float* __restrict__ Wv,
                                                         unsigned short* __restrict__ wcat) {
  int i = blockIdx.x * blockDim.x + threadIdx.x;
  if (i >= NQKV * DD) return;
  int n = i / DD, k = i % DD;
  float v;
  if (n < DD)            v = Wq[(size_t)k * DD + n];
  else if (n < DD + 256) v = Wk[(size_t)k * 256 + (n - DD)];
  else                   v = Wv[(size_t)k * 256 + (n - DD - 256)];
  wcat[(size_t)n * DD + k] = f2bf(v);
}

__global__ void __launch_bounds__(256) build_wot_kernel(const float* __restrict__ Wo,
                                                        unsigned short* __restrict__ wot) {
  int i = blockIdx.x * blockDim.x + threadIdx.x;
  if (i >= DD * DD) return;
  int n = i / DD, k = i % DD;
  wot[(size_t)n * DD + k] = f2bf(Wo[(size_t)k * DD + n]);
}

// ---------------- bf16 WMMA GEMM: C(MxN) = A(MxK) @ B^T(NxK) ----------------
// Per wave: 32(M) x 64(N) tile. Manual 2x ping-pong unroll: fragment set A/B
// alternate, each half-step prefetches into the other set (no register copies).
// Requires K % 64 == 0 (true for K=768).
#define GEMM_WAVES 4
__global__ void __launch_bounds__(32 * GEMM_WAVES, 1)
gemm_bf16_kernel(const unsigned short* __restrict__ A,
                 const unsigned short* __restrict__ Bt,
                 float* __restrict__ C, int M, int N, int K) {
  int lane = threadIdx.x & 31;
  int wave = threadIdx.x >> 5;
  int mt = blockIdx.x * GEMM_WAVES + wave;       // 32-row tile index
  int Mt = (M + 31) >> 5;
  if (mt >= Mt) return;                          // wave-uniform
  int n0 = blockIdx.y * 64;
  int half = lane >> 4, mrow = lane & 15;
  int r0 = mt * 32 + mrow;       if (r0 >= M) r0 = M - 1;   // clamp loads
  int r1 = mt * 32 + 16 + mrow;  if (r1 >= M) r1 = M - 1;
  int aoff0 = r0 * K;
  int aoff1 = r1 * K;
  int boff  = n0 * K;

  v8f acc0[4] = {}, acc1[4] = {};
  v16bf a0A, a1A, bA[4], a0B, a1B, bB[4];
  a0A = load_a_frag(A, aoff0, half);
  a1A = load_a_frag(A, aoff1, half);
  #pragma unroll
  for (int nt = 0; nt < 4; ++nt)
    bA[nt] = load_b_frag(Bt, boff + nt * 16 * K, K, lane);

  for (int k0 = 0; k0 < K; k0 += 64) {
    int kB = k0 + 32;                            // < K since K % 64 == 0
    a0B = load_a_frag(A, aoff0 + kB, half);
    a1B = load_a_frag(A, aoff1 + kB, half);
    #pragma unroll
    for (int nt = 0; nt < 4; ++nt)
      bB[nt] = load_b_frag(Bt, boff + nt * 16 * K + kB, K, lane);
    #pragma unroll
    for (int nt = 0; nt < 4; ++nt) {
      acc0[nt] = wmma_bf16(a0A, bA[nt], acc0[nt]);
      acc1[nt] = wmma_bf16(a1A, bA[nt], acc1[nt]);
    }
    int kA = (k0 + 64 < K) ? k0 + 64 : 0;        // clamped redundant last load
    a0A = load_a_frag(A, aoff0 + kA, half);
    a1A = load_a_frag(A, aoff1 + kA, half);
    #pragma unroll
    for (int nt = 0; nt < 4; ++nt)
      bA[nt] = load_b_frag(Bt, boff + nt * 16 * K + kA, K, lane);
    #pragma unroll
    for (int nt = 0; nt < 4; ++nt) {
      acc0[nt] = wmma_bf16(a0B, bB[nt], acc0[nt]);
      acc1[nt] = wmma_bf16(a1B, bB[nt], acc1[nt]);
    }
  }

  int col = lane & 15;
  #pragma unroll
  for (int nt = 0; nt < 4; ++nt)
    #pragma unroll
    for (int r = 0; r < 8; ++r) {
      int row0 = mt * 32 + r + 8 * half;
      int row1 = row0 + 16;
      if (row0 < M) C[(size_t)row0 * N + n0 + nt * 16 + col] = acc0[nt][r];
      if (row1 < M) C[(size_t)row1 * N + n0 + nt * 16 + col] = acc1[nt][r];
    }
}

// ---------------- RMSNorm + 2D-RoPE + layout (Q/K padded, V transposed) -----
__global__ void __launch_bounds__(128) postproc_kernel(const float* __restrict__ qkv,
                                const float* __restrict__ qw,
                                const float* __restrict__ kw,
                                const int* __restrict__ pos,
                                unsigned short* __restrict__ qb,
                                unsigned short* __restrict__ kb,
                                unsigned short* __restrict__ vtb) {
  int lane = threadIdx.x & 31;
  int wave = threadIdx.x >> 5;          // 0..3
  int b = blockIdx.x / SPAD;
  int s = blockIdx.x % SPAD;
  bool in_s = (s < SS);
  bool dorot = in_s && (s >= 1);        // reference skips RoPE at seq pos 0
  float cs = 1.f, sn = 0.f;
  if (dorot) {
    int ids = pos[b * SS + s];
    int hi = (ids / GW_C) % GH_C;
    int wi = ids % GW_C;
    int p = lane, j = p & 15;
    float invf = __expf(-(float)j * (9.21034037198f / 16.f)); // 10000^(-j/16)
    float ang = (p < 16 ? (float)hi : (float)wi) * invf;
    cs = __cosf(ang); sn = __sinf(ang);
  }
  const float* rowbase = qkv + (size_t)(b * SS + s) * NQKV;

  for (int it = 0; it < 3; ++it) {      // Q heads {wave, wave+4, wave+8}
    int hh = wave + it * 4;
    float y0 = 0.f, y1 = 0.f;
    if (in_s) {
      const float* x = rowbase + hh * HDIM;
      float x0 = x[2 * lane], x1 = x[2 * lane + 1];
      float ss2 = x0 * x0 + x1 * x1;
      for (int off = 16; off; off >>= 1) ss2 += __shfl_xor(ss2, off);
      float rn = rsqrtf(ss2 * (1.f / 64.f) + EPSC);
      y0 = x0 * rn * qw[2 * lane];
      y1 = x1 * rn * qw[2 * lane + 1];
      if (dorot) { float o0 = y0 * cs - y1 * sn, o1 = y0 * sn + y1 * cs; y0 = o0; y1 = o1; }
    }
    unsigned short* dst = qb + ((size_t)(b * NH + hh) * SPAD + s) * HDIM;
    dst[2 * lane] = f2bf(y0); dst[2 * lane + 1] = f2bf(y1);
  }
  {                                      // K head = wave
    float y0 = 0.f, y1 = 0.f;
    if (in_s) {
      const float* x = rowbase + DD + wave * HDIM;
      float x0 = x[2 * lane], x1 = x[2 * lane + 1];
      float ss2 = x0 * x0 + x1 * x1;
      for (int off = 16; off; off >>= 1) ss2 += __shfl_xor(ss2, off);
      float rn = rsqrtf(ss2 * (1.f / 64.f) + EPSC);
      y0 = x0 * rn * kw[2 * lane];
      y1 = x1 * rn * kw[2 * lane + 1];
      if (dorot) { float o0 = y0 * cs - y1 * sn, o1 = y0 * sn + y1 * cs; y0 = o0; y1 = o1; }
    }
    unsigned short* dst = kb + ((size_t)(b * NKV + wave) * SPAD + s) * HDIM;
    dst[2 * lane] = f2bf(y0); dst[2 * lane + 1] = f2bf(y1);
  }
  {                                      // V head = wave, stored HD x SPAD
    float y0 = 0.f, y1 = 0.f;
    if (in_s) {
      const float* x = rowbase + DD + 256 + wave * HDIM;
      y0 = x[2 * lane]; y1 = x[2 * lane + 1];
    }
    unsigned short* dst = vtb + (size_t)(b * NKV + wave) * HDIM * SPAD;
    dst[(size_t)(2 * lane) * SPAD + s]     = f2bf(y0);
    dst[(size_t)(2 * lane + 1) * SPAD + s] = f2bf(y1);
  }
}

// ---------------- attention: one wave per (b, h, 16-query tile) -------------
// LDS layout: [0,2KB) and [2KB,4KB) = TDM double-buffered K tiles;
//             [4KB, 4KB+16*SPAD*4) fp32 scores; then 16*SPAD*2 bf16 probs.
__global__ void __launch_bounds__(32, 1)
attn_kernel(const unsigned short* __restrict__ qb,
            const unsigned short* __restrict__ kb,
            const unsigned short* __restrict__ vtb,
            const unsigned char* __restrict__ mask,
            float* __restrict__ attn_out,
            unsigned short* __restrict__ Ob) {
  extern __shared__ char smem[];
  unsigned short* kt0 = (unsigned short*)smem;
  unsigned short* kt1 = (unsigned short*)(smem + 2048);
  float*          Ps  = (float*)(smem + 4096);
  unsigned short* Pb  = (unsigned short*)(smem + 4096 + 16 * SPAD * 4);

  int lane = threadIdx.x & 31;
  int qt = blockIdx.x;     // 0..64
  int h  = blockIdx.y;     // 0..11
  int b  = blockIdx.z;     // 0..7
  int kvh = h / NREP;
  int half = lane >> 4, mrow = lane & 15;

  int qoff = ((b * NH + h) * SPAD + qt * 16) * HDIM;   // < 6.5M, fits i32
  const unsigned short* kbase = kb + (size_t)(b * NKV + kvh) * SPAD * HDIM;

  v16bf a0 = load_a_frag(qb, qoff + mrow * HDIM, half);        // hd 0..31
  v16bf a1 = load_a_frag(qb, qoff + mrow * HDIM + 32, half);   // hd 32..63

  // ---- scores: TDM-pipelined K tiles (each 16 keys x 64 hd = contiguous 2KB)
  // 2x unrolled so the two LDS buffers alternate statically.
  tdm_load_2kb(kbase, kt0);
  for (int kt = 0; kt < NKT; kt += 2) {
    // even tile (in kt0); prefetch odd tile into kt1
    asm volatile("" ::: "memory");
    tdm_load_2kb(kbase + (kt + 1) * 16 * HDIM, kt1);
    __builtin_amdgcn_s_wait_tensorcnt(1);
    {
      v16bf b0 = load_b_frag(kt0, 0, HDIM, lane);
      v16bf b1 = load_b_frag(kt0, 32, HDIM, lane);
      v8f acc = {};
      acc = wmma_bf16(a0, b0, acc);
      acc = wmma_bf16(a1, b1, acc);
      int col = kt * 16 + (lane & 15);
      bool valid = (col < SS) && (mask[b * SS + col] != 0);
      float addv = valid ? 0.f : -3.0e38f;
      #pragma unroll
      for (int r = 0; r < 8; ++r)
        Ps[(r + 8 * half) * SPAD + col] = acc[r] * 0.125f + addv;
    }
    // odd tile (in kt1); prefetch next even tile into kt0
    int kt2 = (kt + 2 < NKT) ? kt + 2 : 0;
    asm volatile("" ::: "memory");
    tdm_load_2kb(kbase + kt2 * 16 * HDIM, kt0);
    __builtin_amdgcn_s_wait_tensorcnt(1);
    {
      v16bf b0 = load_b_frag(kt1, 0, HDIM, lane);
      v16bf b1 = load_b_frag(kt1, 32, HDIM, lane);
      v8f acc = {};
      acc = wmma_bf16(a0, b0, acc);
      acc = wmma_bf16(a1, b1, acc);
      int col = (kt + 1) * 16 + (lane & 15);
      bool valid = (col < SS) && (mask[b * SS + col] != 0);
      float addv = valid ? 0.f : -3.0e38f;
      #pragma unroll
      for (int r = 0; r < 8; ++r)
        Ps[(r + 8 * half) * SPAD + col] = acc[r] * 0.125f + addv;
    }
  }
  __builtin_amdgcn_s_wait_tensorcnt(0);
  __syncthreads();

  // ---- row softmax (wave-cooperative), write fp32 attn + bf16 probs
  for (int r = 0; r < 16; ++r) {
    float m = -3.4e38f;
    for (int c = lane; c < SPAD; c += 32) m = fmaxf(m, Ps[r * SPAD + c]);
    for (int off = 16; off; off >>= 1) m = fmaxf(m, __shfl_xor(m, off));
    float sum = 0.f;
    for (int c = lane; c < SPAD; c += 32) {
      float e = __expf(Ps[r * SPAD + c] - m);
      Ps[r * SPAD + c] = e;
      sum += e;
    }
    for (int off = 16; off; off >>= 1) sum += __shfl_xor(sum, off);
    float inv = 1.f / sum;
    int qrow = qt * 16 + r;
    for (int c = lane; c < SPAD; c += 32) {
      float p = Ps[r * SPAD + c] * inv;
      Pb[r * SPAD + c] = f2bf(p);
      if (qrow < SS && c < SS)
        attn_out[((size_t)(b * NH + h) * SS + qrow) * SS + c] = p;
    }
  }
  __syncthreads();

  // ---- out tile: P (16 x SPAD) @ V (SPAD x 64); ping-pong + tail (33 steps)
  int voff = (b * NKV + kvh) * HDIM * SPAD;      // < 2.2M, fits i32
  v8f o[4] = {};
  v16bf apA, bvA[4], apB, bvB[4];
  apA = load_a_frag(Pb, mrow * SPAD, half);
  #pragma unroll
  for (int nt = 0; nt < 4; ++nt)
    bvA[nt] = load_b_frag(vtb, voff + nt * 16 * SPAD, SPAD, lane);
  for (int ks = 0; ks < SPAD - 32; ks += 64) {   // 16 double-steps (ks<1024)
    apB = load_a_frag(Pb, mrow * SPAD + ks + 32, half);
    #pragma unroll
    for (int nt = 0; nt < 4; ++nt)
      bvB[nt] = load_b_frag(vtb, voff + nt * 16 * SPAD + ks + 32, SPAD, lane);
    #pragma unroll
    for (int nt = 0; nt < 4; ++nt)
      o[nt] = wmma_bf16(apA, bvA[nt], o[nt]);
    apA = load_a_frag(Pb, mrow * SPAD + ks + 64, half);      // ks+64 <= 1024
    #pragma unroll
    for (int nt = 0; nt < 4; ++nt)
      bvA[nt] = load_b_frag(vtb, voff + nt * 16 * SPAD + ks + 64, SPAD, lane);
    #pragma unroll
    for (int nt = 0; nt < 4; ++nt)
      o[nt] = wmma_bf16(apB, bvB[nt], o[nt]);
  }
  #pragma unroll
  for (int nt = 0; nt < 4; ++nt)                 // tail: ks = 1024 (set A)
    o[nt] = wmma_bf16(apA, bvA[nt], o[nt]);

  int col = lane & 15;
  #pragma unroll
  for (int nt = 0; nt < 4; ++nt)
    #pragma unroll
    for (int r = 0; r < 8; ++r) {
      int qrow = qt * 16 + r + 8 * half;
      if (qrow < SS)
        Ob[(size_t)(b * SS + qrow) * DD + h * HDIM + nt * 16 + col] = f2bf(o[nt][r]);
    }
}

// ---------------- host-side orchestration -----------------------------------
extern "C" void kernel_launch(void* const* d_in, const int* in_sizes, int n_in,
                              void* d_out, int out_size, void* d_ws, size_t ws_size,
                              hipStream_t stream) {
  const float*         hs   = (const float*)d_in[0];
  const unsigned char* mask = (const unsigned char*)d_in[1];
  const int*           pos  = (const int*)d_in[2];
  const float*         Wq   = (const float*)d_in[5];
  const float*         Wk   = (const float*)d_in[6];
  const float*         Wv   = (const float*)d_in[7];
  const float*         Wo   = (const float*)d_in[8];
  const float*         qw   = (const float*)d_in[9];
  const float*         kw   = (const float*)d_in[10];

  const int M = BB * SS;                 // 8200 rows
  char* ws = (char*)d_ws;
  size_t off = 0;
  unsigned short* hsb  = (unsigned short*)(ws + off); off += (size_t)M * DD * 2;
  unsigned short* wcat = (unsigned short*)(ws + off); off += (size_t)NQKV * DD * 2;
  unsigned short* wot  = (unsigned short*)(ws + off); off += (size_t)DD * DD * 2;
  float*          qkv  = (float*)(ws + off);          off += (size_t)M * NQKV * 4;
  unsigned short* qb   = (unsigned short*)(ws + off); off += (size_t)BB * NH  * SPAD * HDIM * 2;
  unsigned short* kb   = (unsigned short*)(ws + off); off += (size_t)BB * NKV * SPAD * HDIM * 2;
  unsigned short* vtb  = (unsigned short*)(ws + off); off += (size_t)BB * NKV * HDIM * SPAD * 2;
  unsigned short* Ob   = (unsigned short*)(ws + off); off += (size_t)M * DD * 2;
  // total ~92 MB of workspace

  float* outp  = (float*)d_out;                       // (B, S, D) fp32
  float* attnp = outp + (size_t)BB * SS * DD;         // (B, H, S, S) fp32

  { // 1) hidden -> bf16
    int n = M * DD;
    cvt_bf16_kernel<<<(n + 255) / 256, 256, 0, stream>>>(hs, hsb, n);
  }
  { // 2) transposed bf16 weights
    int n = NQKV * DD;
    build_wcat_kernel<<<(n + 255) / 256, 256, 0, stream>>>(Wq, Wk, Wv, wcat);
    int n2 = DD * DD;
    build_wot_kernel<<<(n2 + 255) / 256, 256, 0, stream>>>(Wo, wot);
  }
  { // 3) QKV = X @ Wcat  (M=8200, N=1280, K=768)
    int Mt = (M + 31) / 32;                                  // 257
    dim3 grid((Mt + GEMM_WAVES - 1) / GEMM_WAVES, NQKV / 64);
    gemm_bf16_kernel<<<grid, 32 * GEMM_WAVES, 0, stream>>>(hsb, wcat, qkv, M, NQKV, DD);
  }
  // 4) RMSNorm + RoPE + layout
  postproc_kernel<<<BB * SPAD, 128, 0, stream>>>(qkv, qw, kw, pos, qb, kb, vtb);
  { // 5) attention (softmax materialized into d_out's attn region)
    dim3 grid((SS + 15) / 16, NH, BB);                       // (65, 12, 8)
    size_t lds = 4096 + (size_t)16 * SPAD * 4 + (size_t)16 * SPAD * 2; // 105472 B
    attn_kernel<<<grid, 32, lds, stream>>>(qb, kb, vtb, mask, attnp, Ob);
  }
  { // 6) out = O @ Wo  (M=8200, N=768, K=768)
    int Mt = (M + 31) / 32;
    dim3 grid((Mt + GEMM_WAVES - 1) / GEMM_WAVES, DD / 64);
    gemm_bf16_kernel<<<grid, 32 * GEMM_WAVES, 0, stream>>>(Ob, wot, outp, M, DD, DD);
  }
  (void)in_sizes; (void)n_in; (void)out_size; (void)ws_size;
}
